// multipole_interaction_9216999817545
// MI455X (gfx1250) — matrix operand order repaired and checked by reading, not compiled, not run on hardware
//
#include <hip/hip_runtime.h>
#include <hip/hip_bf16.h>

typedef __attribute__((ext_vector_type(16))) __bf16 v16bf;
typedef __attribute__((ext_vector_type(8)))  __bf16 v8bf;
typedef __attribute__((ext_vector_type(8)))  float  v8f;

#define FDIM 128
#define ADIM 128
#define TILE 16
#define XSTRIDE 136       // bf16 elems per LDS row: 128 + 8 pad (row = 272B -> bank stride 4, conflict-free)
#define EPSV 1e-7f
#define NMAT 8            // Wq1, Wq2, WT1, WT2, WT3, Ws1, Ws2, Wv
#define MAT_ELEMS (FDIM * FDIM)

__device__ __forceinline__ float fast_rcp(float x)  { return __builtin_amdgcn_rcpf(x); }
__device__ __forceinline__ float fast_sqrt(float x) { return __builtin_amdgcn_sqrtf(x); }

__device__ __forceinline__ float swishf(float x) {
  // x * sigmoid(x) with hw rcp (v_exp_f32 + v_rcp_f32), avoids IEEE div_scale/div_fmas chain
  return x * fast_rcp(1.0f + __expf(-x));
}

__device__ __forceinline__ v8f vzero() {
  v8f z;
#pragma unroll
  for (int e = 0; e < 8; ++e) z[e] = 0.0f;
  return z;
}

// ---------------- weight pre-conversion: f32 -> bf16 in B-fragment order ----------------
// Layout: Wb[(((mat*8 + ct)*4 + kb)*32 + lane)*16 + e] = W[(kb*32 + (lane>>4)*16 + e)*128 + ct*16 + (lane&15)]
__global__ __launch_bounds__(256)
void weights_to_bf16_kernel(const float* __restrict__ Wq1, const float* __restrict__ Wq2,
                            const float* __restrict__ WT1, const float* __restrict__ WT2,
                            const float* __restrict__ WT3, const float* __restrict__ Ws1,
                            const float* __restrict__ Ws2, const float* __restrict__ Wv,
                            __bf16* __restrict__ Wb) {
  int idx = blockIdx.x * 256 + threadIdx.x;
  if (idx >= NMAT * MAT_ELEMS) return;
  int mat  = idx >> 14;
  int rem  = idx & (MAT_ELEMS - 1);
  int e    = rem & 15;
  int lane = (rem >> 4) & 31;
  int kb   = (rem >> 9) & 3;
  int ct   = rem >> 11;
  const float* W;
  switch (mat) {
    case 0: W = Wq1; break;
    case 1: W = Wq2; break;
    case 2: W = WT1; break;
    case 3: W = WT2; break;
    case 4: W = WT3; break;
    case 5: W = Ws1; break;
    case 6: W = Ws2; break;
    default: W = Wv; break;
  }
  int k   = kb * 32 + (lane >> 4) * 16 + e;
  int col = ct * 16 + (lane & 15);
  Wb[idx] = (__bf16)W[k * FDIM + col];
}

// ---------------- per-wave B fragments held in VGPRs ----------------
struct BFrags { v16bf f[4]; };

__device__ __forceinline__ BFrags load_bfrags(const __bf16* __restrict__ Wb,
                                              int mat, int ct, int lane) {
  BFrags r;
#pragma unroll
  for (int kb = 0; kb < 4; ++kb)
    r.f[kb] = *(const v16bf*)(Wb + ((((size_t)mat * 8 + ct) * 4 + kb) * 32 + lane) * 16);
  return r;
}

// acc += X(16xF bf16, LDS) @ W columns [ct*16, ct*16+16) with preloaded B fragments
__device__ __forceinline__ v8f gemm_tile(const __bf16* __restrict__ Xlds,
                                         const BFrags& B, int lane, v8f acc) {
  const int r    = lane & 15;
  const int half = lane >> 4;
#pragma unroll
  for (int kb = 0; kb < 4; ++kb) {
    const __bf16* abase = Xlds + r * XSTRIDE + kb * 32 + half * 16;
    v8bf alo = *(const v8bf*)(abase);        // K = kb*32 + half*16 + [0..7]
    v8bf ahi = *(const v8bf*)(abase + 8);    // K = kb*32 + half*16 + 8 + [0..7]
    v16bf a = __builtin_shufflevector(alo, ahi,
                0, 1, 2, 3, 4, 5, 6, 7, 8, 9, 10, 11, 12, 13, 14, 15);
    acc = __builtin_amdgcn_wmma_f32_16x16x32_bf16(false, a, false, B.f[kb], (short)0,
                                                  acc, false, false);
  }
  return acc;
}

// write C-layout v8f into LDS bf16 tile (row = half*8+v, col = ct*16 + lane&15)
__device__ __forceinline__ void store_tile_bf16(__bf16* Xlds, int ct, int lane, v8f vals) {
  const int c    = lane & 15;
  const int half = lane >> 4;
#pragma unroll
  for (int v = 0; v < 8; ++v)
    Xlds[(half * 8 + v) * XSTRIDE + ct * 16 + c] = (__bf16)vals[v];
}

__global__ __launch_bounds__(256)
void multipole_fused_kernel(const float* __restrict__ feat, const float* __restrict__ mu,
                            const float* __restrict__ T0, const float* __restrict__ T1,
                            const float* __restrict__ T2, const float* __restrict__ T3,
                            const __bf16* __restrict__ Wb,
                            const float* __restrict__ bq1, const float* __restrict__ bq2,
                            const float* __restrict__ bs1, const float* __restrict__ bs2,
                            const float* __restrict__ cM1, const float* __restrict__ cM2,
                            const float* __restrict__ cM3,
                            float* __restrict__ ms_out, float* __restrict__ mv_out,
                            int n_atoms) {
  __shared__ __bf16 X[TILE * XSTRIDE];
  __shared__ float  Tsh[TILE * 40];   // per row: [0]=T0, [1..3]=T1, [4..12]=T2, [13..39]=T3

  const int tid  = threadIdx.x;
  const int lane = tid & 31;
  const int ct   = tid >> 5;          // wave id = column tile (0..7)
  const int half = lane >> 4;
  const int c    = lane & 15;
  const int base = blockIdx.x * TILE;

  // prefetch next tile's streaming inputs (global_prefetch_b8)
  if (base + TILE < n_atoms) {
    __builtin_prefetch(feat + (size_t)(base + TILE) * FDIM, 0, 0);
    __builtin_prefetch(mu + (size_t)(base + TILE) * FDIM * 3, 0, 0);
  }

  // ---- cooperative load: multipole tensors into LDS ----
  for (int idx = tid; idx < TILE * 40; idx += 256) {
    int row = idx / 40, j = idx % 40;
    int n = base + row; if (n >= n_atoms) n = n_atoms - 1;
    float vld;
    if (j == 0)       vld = T0[n];
    else if (j < 4)   vld = T1[n * 3 + (j - 1)];
    else if (j < 13)  vld = T2[n * 9 + (j - 4)];
    else              vld = T3[n * 27 + (j - 13)];
    Tsh[row * 40 + j] = vld;
  }
  // ---- feat tile -> X (bf16) ----
  for (int idx = tid; idx < TILE * FDIM; idx += 256) {
    int row = idx >> 7, colf = idx & 127;
    int n = base + row; if (n >= n_atoms) n = n_atoms - 1;
    X[row * XSTRIDE + colf] = (__bf16)feat[(size_t)n * FDIM + colf];
  }
  __syncthreads();

  const int fcol = ct * 16 + c;       // this lane's output feature column

  // ================= M0 = swish(feat@Wq1+bq1)@Wq2 + bq2 =================
  v8f M0;
  {
    BFrags Bq1 = load_bfrags(Wb, 0, ct, lane);
    v8f acc = gemm_tile(X, Bq1, lane, vzero());
    float b1 = bq1[fcol];
    v8f H;
#pragma unroll
    for (int v = 0; v < 8; ++v) H[v] = swishf(acc[v] + b1);
    __syncthreads();
    store_tile_bf16(X, ct, lane, H);
    __syncthreads();
    BFrags Bq2 = load_bfrags(Wb, 1, ct, lane);
    M0 = gemm_tile(X, Bq2, lane, vzero());
    float b2 = bq2[fcol];
#pragma unroll
    for (int v = 0; v < 8; ++v) M0[v] += b2;
  }

  // ================= projections M1/M2/M3 for each spatial component =================
  v8f M1[3], M2[3], M3[3];
  {
    BFrags BT1 = load_bfrags(Wb, 2, ct, lane);   // reused across all 3 components
    BFrags BT2 = load_bfrags(Wb, 3, ct, lane);
    BFrags BT3 = load_bfrags(Wb, 4, ct, lane);
#pragma unroll
    for (int i = 0; i < 3; ++i) {
      __syncthreads();
      for (int idx = tid; idx < TILE * FDIM; idx += 256) {
        int row = idx >> 7, colf = idx & 127;
        int n = base + row; if (n >= n_atoms) n = n_atoms - 1;
        X[row * XSTRIDE + colf] = (__bf16)mu[((size_t)n * FDIM + colf) * 3 + i];
      }
      __syncthreads();
      M1[i] = gemm_tile(X, BT1, lane, vzero());
      M2[i] = gemm_tile(X, BT2, lane, vzero());
      M3[i] = gemm_tile(X, BT3, lane, vzero());
    }
  }

  // ---- norm + gating: M = M / (c^2 * |M + EPS| + 1)  (hw sqrt/rcp) ----
  {
    float c1 = cM1[fcol]; c1 *= c1;
    float c2 = cM2[fcol]; c2 *= c2;
    float c3 = cM3[fcol]; c3 *= c3;
#pragma unroll
    for (int v = 0; v < 8; ++v) {
      float a0 = M1[0][v] + EPSV, a1 = M1[1][v] + EPSV, a2 = M1[2][v] + EPSV;
      float d1 = fast_rcp(c1 * fast_sqrt(a0 * a0 + a1 * a1 + a2 * a2) + 1.0f);
      float b0 = M2[0][v] + EPSV, b1 = M2[1][v] + EPSV, b2 = M2[2][v] + EPSV;
      float d2 = fast_rcp(c2 * fast_sqrt(b0 * b0 + b1 * b1 + b2 * b2) + 1.0f);
      float g0 = M3[0][v] + EPSV, g1 = M3[1][v] + EPSV, g2 = M3[2][v] + EPSV;
      float d3 = fast_rcp(c3 * fast_sqrt(g0 * g0 + g1 * g1 + g2 * g2) + 1.0f);
      M1[0][v] *= d1; M1[1][v] *= d1; M1[2][v] *= d1;
      M2[0][v] *= d2; M2[1][v] *= d2; M2[2][v] *= d2;
      M3[0][v] *= d3; M3[1][v] *= d3; M3[2][v] *= d3;
    }
  }

  // ================= multipole contractions (VALU; T* broadcast from LDS) =================
  v8f ms = vzero(), mv0 = vzero(), mv1 = vzero(), mv2 = vzero();
#pragma unroll
  for (int v = 0; v < 8; ++v) {
    const float* tr = &Tsh[(half * 8 + v) * 40];
    float t0 = tr[0];
    float m0 = M0[v];
    float m1[3] = { M1[0][v], M1[1][v], M1[2][v] };
    float m2[3] = { M2[0][v], M2[1][v], M2[2][v] };
    float m3[3] = { M3[0][v], M3[1][v], M3[2][v] };

    float s = t0 * m0;
    float vi[3];
#pragma unroll
    for (int i = 0; i < 3; ++i) {
      float t1i = tr[1 + i];
      s += m1[i] * t1i;           // einsum('nfi,ni->nf', M1, T1)
      vi[i] = t1i * m0;           // T1 ⊗ M0
    }
#pragma unroll
    for (int i = 0; i < 3; ++i) {
#pragma unroll
      for (int j = 0; j < 3; ++j) {
        float t2ij = tr[4 + i * 3 + j];
        s     += t2ij * m2[i] * m2[j];   // T2 : M2⊗M2
        vi[i] += t2ij * m1[j];           // T2 · M1
      }
    }
#pragma unroll
    for (int i = 0; i < 3; ++i) {
#pragma unroll
      for (int j = 0; j < 3; ++j) {
        float pij = m3[i] * m3[j];
        float q2j = m2[j];
#pragma unroll
        for (int k = 0; k < 3; ++k) {
          float t3ijk = tr[13 + (i * 3 + j) * 3 + k];
          s     += t3ijk * pij * m3[k];  // T3 : M3⊗M3⊗M3
          vi[i] += t3ijk * q2j * m2[k];  // T3 : M2⊗M2 -> vector
        }
      }
    }
    ms[v] = s; mv0[v] = vi[0]; mv1[v] = vi[1]; mv2[v] = vi[2];
  }

  // ================= ms_out = swish(ms@Ws1+bs1)@Ws2 + bs2 =================
  __syncthreads();
  store_tile_bf16(X, ct, lane, ms);
  __syncthreads();
  {
    BFrags Bs1 = load_bfrags(Wb, 5, ct, lane);
    v8f h2 = gemm_tile(X, Bs1, lane, vzero());
    float b1 = bs1[fcol];
#pragma unroll
    for (int v = 0; v < 8; ++v) h2[v] = swishf(h2[v] + b1);
    __syncthreads();
    store_tile_bf16(X, ct, lane, h2);
    __syncthreads();
    BFrags Bs2 = load_bfrags(Wb, 6, ct, lane);
    v8f so = gemm_tile(X, Bs2, lane, vzero());
    float b2 = bs2[fcol];
#pragma unroll
    for (int v = 0; v < 8; ++v) {
      int row = base + half * 8 + v;
      if (row < n_atoms) ms_out[(size_t)row * ADIM + fcol] = so[v] + b2;
    }
  }

  // ================= mv_out[:, :, i] = mv_i @ Wv =================
  {
    BFrags Bv = load_bfrags(Wb, 7, ct, lane);    // reused across all 3 components
#pragma unroll
    for (int i = 0; i < 3; ++i) {
      v8f mvi = (i == 0) ? mv0 : ((i == 1) ? mv1 : mv2);
      __syncthreads();
      store_tile_bf16(X, ct, lane, mvi);
      __syncthreads();
      v8f vo = gemm_tile(X, Bv, lane, vzero());
#pragma unroll
      for (int v = 0; v < 8; ++v) {
        int row = base + half * 8 + v;
        if (row < n_atoms) mv_out[((size_t)row * ADIM + fcol) * 3 + i] = vo[v];
      }
    }
  }
}

extern "C" void kernel_launch(void* const* d_in, const int* in_sizes, int n_in,
                              void* d_out, int out_size, void* d_ws, size_t ws_size,
                              hipStream_t stream) {
  const float* feat = (const float*)d_in[0];
  const float* mu   = (const float*)d_in[1];
  const float* T0   = (const float*)d_in[2];
  const float* T1   = (const float*)d_in[3];
  const float* T2   = (const float*)d_in[4];
  const float* T3   = (const float*)d_in[5];
  const float* Wq1  = (const float*)d_in[6];
  const float* bq1  = (const float*)d_in[7];
  const float* Wq2  = (const float*)d_in[8];
  const float* bq2  = (const float*)d_in[9];
  const float* WT1  = (const float*)d_in[10];
  const float* WT2  = (const float*)d_in[11];
  const float* WT3  = (const float*)d_in[12];
  const float* Ws1  = (const float*)d_in[13];
  const float* bs1  = (const float*)d_in[14];
  const float* Ws2  = (const float*)d_in[15];
  const float* bs2  = (const float*)d_in[16];
  const float* Wv   = (const float*)d_in[17];
  const float* cM1  = (const float*)d_in[18];
  const float* cM2  = (const float*)d_in[19];
  const float* cM3  = (const float*)d_in[20];

  const int n_atoms = in_sizes[0] / FDIM;
  float* ms_out = (float*)d_out;
  float* mv_out = (float*)d_out + (size_t)n_atoms * ADIM;

  __bf16* Wb = (__bf16*)d_ws;   // 8 * 128 * 128 * 2B = 256 KB of scratch

  // 1) pre-convert weights into bf16 B-fragment layout
  const int prep_grid = (NMAT * MAT_ELEMS + 255) / 256;
  hipLaunchKernelGGL(weights_to_bf16_kernel, dim3(prep_grid), dim3(256), 0, stream,
                     Wq1, Wq2, WT1, WT2, WT3, Ws1, Ws2, Wv, Wb);

  // 2) fused multipole interaction
  const int grid = (n_atoms + TILE - 1) / TILE;
  hipLaunchKernelGGL(multipole_fused_kernel, dim3(grid), dim3(256), 0, stream,
                     feat, mu, T0, T1, T2, T3, Wb,
                     bq1, bq2, bs1, bs2, cM1, cM2, cM3,
                     ms_out, mv_out, n_atoms);
}
